// EquivariantDecoder_32263794328395
// MI455X (gfx1250) — matrix operand order repaired and checked by reading, not compiled, not run on hardware
//
#include <hip/hip_runtime.h>

#define LAYERS   4
#define NODE_DIM 128
#define HIDDEN   256
#define EDGE_DIM 64
#define BATCH    2
#define NN       192
#define NT       12      // NN / 16
#define CUTOFF_R 2.0f

typedef __attribute__((ext_vector_type(16))) __bf16 v16bf;
typedef __attribute__((ext_vector_type(8)))  float  v8f;

__device__ __forceinline__ float silu_f(float x) {
  // x * sigmoid(x); v_rcp_f32 instead of the IEEE divide expansion
  return x * __builtin_amdgcn_rcpf(1.0f + __expf(-x));
}

// ---------------- workspace layout (float element offsets) ----------------
#define OFF_XCUR   0                                    // B*N*3
#define OFF_HCUR   (OFF_XCUR  + BATCH*NN*3)             // B*N*128
#define OFF_D      (OFF_HCUR  + BATCH*NN*NODE_DIM)      // B*N*N  [b][s][d]
#define OFF_MASK   (OFF_D     + BATCH*NN*NN)            // B*N*N
#define OFF_CNT    (OFF_MASK  + BATCH*NN*NN)            // B*N
#define OFF_PRE1N  (OFF_CNT   + BATCH*NN)               // B*N*256
#define OFF_PRE2N  (OFF_PRE1N + BATCH*NN*HIDDEN)
#define OFF_PRE1C  (OFF_PRE2N + BATCH*NN*HIDDEN)
#define OFF_PRE2C  (OFF_PRE1C + BATCH*NN*HIDDEN)
#define OFF_BIASN  (OFF_PRE2C + BATCH*NN*HIDDEN)        // L*256
#define OFF_BIASC  (OFF_BIASN + LAYERS*HIDDEN)
#define OFF_AGGN   (OFF_BIASC + LAYERS*HIDDEN)          // B*N*256
#define OFF_PARTN  (OFF_AGGN  + BATCH*NN*HIDDEN)        // B*NT*N*256
#define OFF_PARTC  (OFF_PARTN + BATCH*NT*NN*HIDDEN)     // B*NT*N*3
#define F_TOTAL    (OFF_PARTC + BATCH*NT*NN*3)

// ---------------------------------------------------------------------------
// Prep: fold edge MLP 2nd matmul into node/coord W1 edge-columns, pre-swizzled
// into the WMMA 16-bit B-matrix (32x16) fragment layout. Also total biases.
// Bn[e,h] = sum_j ew2[j,e] * W1[h, 256+j];  biasT[h] = b1[h] + sum_j eb2[j]*W1[h,256+j]
// ---------------------------------------------------------------------------
__global__ void k_prep(const float* __restrict__ ew2, const float* __restrict__ eb2,
                       const float* __restrict__ nw1, const float* __restrict__ nb1,
                       const float* __restrict__ cw1, const float* __restrict__ cb1,
                       unsigned short* __restrict__ anf_u, unsigned short* __restrict__ acf_u,
                       float* __restrict__ biasn, float* __restrict__ biasc) {
  int tid = blockIdx.x * blockDim.x + threadIdx.x;      // 4*2*64*256 = 131072
  if (tid >= LAYERS * 2 * EDGE_DIM * HIDDEN) return;
  int h = tid & 255;
  int e = (tid >> 8) & 63;
  int p = (tid >> 14) & 1;
  int l = tid >> 15;
  const float* W1 = p ? cw1 : nw1;
  const float* wrow = W1 + (size_t)(l * HIDDEN + h) * 320 + 256;   // edge columns
  const float* e2 = ew2 + (size_t)l * 64 * 64 + e;                 // stride 64 over j
  float val = 0.f;
  for (int j = 0; j < 64; ++j) val += e2[j * 64] * wrow[j];
  // B-fragment position: K = e&31 within kfrag e>>5; lane = n + 16*((K>>3)&1); idx = (K&7) + 8*(K>>4)
  int chunk = h >> 4, n = h & 15, kf = e >> 5, K = e & 31;
  int lane = n + 16 * ((K >> 3) & 1);
  __bf16* dst = (__bf16*)(p ? acf_u : anf_u);
  dst[(size_t)l * 16384 + (((chunk * 2 + kf) * 32 + lane) * 16 + ((K & 7) + 8 * (K >> 4)))] = (__bf16)val;
  if (e == 0) {
    const float* b2 = eb2 + l * 64;
    float bv = 0.f;
    for (int j = 0; j < 64; ++j) bv += b2[j] * wrow[j];
    if (p) biasc[l * HIDDEN + h] = bv + cb1[l * HIDDEN + h];
    else   biasn[l * HIDDEN + h] = bv + nb1[l * HIDDEN + h];
  }
}

// ------------------------- init coords + copy h ---------------------------
__global__ void k_init(const float* __restrict__ h, const float* __restrict__ iw,
                       const float* __restrict__ ib,
                       float* __restrict__ xcur, float* __restrict__ hcur) {
  int t = blockIdx.x * blockDim.x + threadIdx.x;
  if (t >= BATCH * NN) return;
  const float* hr = h + (size_t)t * NODE_DIM;
  float* hw = hcur + (size_t)t * NODE_DIM;
  for (int k = 0; k < NODE_DIM; ++k) hw[k] = hr[k];
  for (int c = 0; c < 3; ++c) {
    float a = ib[c];
    for (int k = 0; k < NODE_DIM; ++k) a += hr[k] * iw[c * NODE_DIM + k];
    xcur[t * 3 + c] = a;
  }
}

// ------------------------- distances + mask -------------------------------
__global__ void k_dist(const float* __restrict__ xcur,
                       float* __restrict__ Dm, float* __restrict__ Mm) {
  int t = blockIdx.x * blockDim.x + threadIdx.x;
  if (t >= BATCH * NN * NN) return;
  int d = t % NN; int s = (t / NN) % NN; int b = t / (NN * NN);
  const float* xs = xcur + ((size_t)b * NN + s) * 3;
  const float* xd = xcur + ((size_t)b * NN + d) * 3;
  float dx = xs[0] - xd[0], dy = xs[1] - xd[1], dz = xs[2] - xd[2];
  float sq = dx * dx + dy * dy + dz * dz;
  float Dv = sq > 0.f ? sqrtf(sq) : 0.f;
  Dm[t] = Dv;
  Mm[t] = (Dv < CUTOFF_R && s != d) ? 1.f : 0.f;
}

__global__ void k_cnt(const float* __restrict__ Mm, float* __restrict__ cnt) {
  int t = blockIdx.x * blockDim.x + threadIdx.x;
  if (t >= BATCH * NN) return;
  int d = t % NN; int b = t / NN;
  float a = 0.f;
  for (int s = 0; s < NN; ++s) a += Mm[((size_t)b * NN + s) * NN + d];
  cnt[t] = a;
}

// --------- per-layer per-node pre-activations (src/dst halves of W1) -------
__global__ void k_pre(int l, const float* __restrict__ hcur,
                      const float* __restrict__ nw1, const float* __restrict__ cw1,
                      float* __restrict__ pre1n, float* __restrict__ pre2n,
                      float* __restrict__ pre1c, float* __restrict__ pre2c) {
  int t = blockIdx.x * blockDim.x + threadIdx.x;       // 4*B*N*256 = 393216
  int hh = t & 255;
  int node = (t >> 8) % NN;
  int rest = (t >> 8) / NN;
  int b = rest & 1;
  int which = rest >> 1;
  if (which >= 4) return;
  const float* W = (which < 2) ? nw1 : cw1;
  int koff = (which & 1) * NODE_DIM;
  const float* wrow = W + (size_t)(l * HIDDEN + hh) * 320 + koff;
  const float* hr = hcur + (size_t)(b * NN + node) * NODE_DIM;
  float a = 0.f;
  for (int k = 0; k < NODE_DIM; ++k) a += hr[k] * wrow[k];
  float* dst = (which == 0) ? pre1n : (which == 1) ? pre2n : (which == 2) ? pre1c : pre2c;
  dst[((size_t)b * NN + node) * HIDDEN + hh] = a;
}

// ---------------------------------------------------------------------------
// Pair kernel: 16x16 (src x dst) pair tile per workgroup; 16 waves, each wave
// owns one dst column. GEMM1 (K=64) via v_wmma_f32_16x16x32_bf16; silu + mask
// + M-reduction in hidden domain; coord path reduced against cw2 per row.
// ---------------------------------------------------------------------------
__global__ __launch_bounds__(512)
void k_pair(int l, const float* __restrict__ Dm, const float* __restrict__ Mm,
            const float* __restrict__ xcur,
            const float* __restrict__ pre1n, const float* __restrict__ pre2n,
            const float* __restrict__ pre1c, const float* __restrict__ pre2c,
            const float* __restrict__ biasn, const float* __restrict__ biasc,
            const unsigned short* __restrict__ anf_u, const unsigned short* __restrict__ acf_u,
            const float* __restrict__ ew1, const float* __restrict__ eb1,
            const float* __restrict__ cw2,
            float* __restrict__ partN, float* __restrict__ partC) {
  __shared__ float p1n[16][HIDDEN];
  __shared__ float p1c[16][HIDDEN];
  int sx = blockIdx.x, dx = blockIdx.y, b = blockIdx.z;
  int s_base = sx * 16, d_base = dx * 16;
  int tid = threadIdx.x;
  for (int i = tid; i < 16 * HIDDEN; i += 512) {
    int s = i >> 8, hh = i & 255;
    p1n[s][hh] = pre1n[((size_t)b * NN + s_base + s) * HIDDEN + hh];
    p1c[s][hh] = pre1c[((size_t)b * NN + s_base + s) * HIDDEN + hh];
  }
  __syncthreads();

  int w = tid >> 5;            // d_local == wave id
  int lane = tid & 31;
  int nn = lane & 15;
  int ha = lane >> 4;
  int d = d_base + w;

  // ---- A fragments: g[s, e] = silu(D[s,d]*ew1[e] + eb1[e]), 16x32 bf16 x2 ----
  float Dv = Dm[((size_t)b * NN + (s_base + nn)) * NN + d];   // row M = lane&15
  const float* ew1l = ew1 + l * EDGE_DIM;
  const float* eb1l = eb1 + l * EDGE_DIM;
  v16bf ga0, ga1;
#pragma unroll
  for (int idx = 0; idx < 16; ++idx) {
    int v = idx >> 1, jb = idx & 1;
    int K = (v < 4) ? (8 * ha + 2 * v + jb) : (16 + 8 * ha + 2 * (v - 4) + jb);
    ga0[idx] = (__bf16)silu_f(Dv * ew1l[K] + eb1l[K]);
    ga1[idx] = (__bf16)silu_f(Dv * ew1l[32 + K] + eb1l[32 + K]);
  }

  float maskr[8];
#pragma unroll
  for (int i = 0; i < 8; ++i)
    maskr[i] = Mm[((size_t)b * NN + (s_base + i + 8 * ha)) * NN + d];

  const __bf16* anf = (const __bf16*)anf_u + (size_t)l * 16384;
  const __bf16* acf = (const __bf16*)acf_u + (size_t)l * 16384;
  const float* bnl  = biasn + l * HIDDEN;
  const float* bcl  = biasc + l * HIDDEN;
  const float* cw2l = cw2   + l * HIDDEN;
  const float* p2n  = pre2n + ((size_t)b * NN + d) * HIDDEN;
  const float* p2c  = pre2c + ((size_t)b * NN + d) * HIDDEN;
  float* pN = partN + (((size_t)b * NT + sx) * NN + d) * HIDDEN;

  float cwacc[8];
#pragma unroll
  for (int i = 0; i < 8; ++i) cwacc[i] = 0.f;

  for (int c16 = 0; c16 < 16; ++c16) {
    int off = c16 * 16;
    v16bf bn0 = *(const v16bf*)(anf + (((c16 * 2 + 0) * 32 + lane) * 16));
    v16bf bn1 = *(const v16bf*)(anf + (((c16 * 2 + 1) * 32 + lane) * 16));
    v8f accn = {0.f, 0.f, 0.f, 0.f, 0.f, 0.f, 0.f, 0.f};
    accn = __builtin_amdgcn_wmma_f32_16x16x32_bf16(false, ga0, false, bn0, (short)0, accn, false, false);
    accn = __builtin_amdgcn_wmma_f32_16x16x32_bf16(false, ga1, false, bn1, (short)0, accn, false, false);
    v16bf bc0 = *(const v16bf*)(acf + (((c16 * 2 + 0) * 32 + lane) * 16));
    v16bf bc1 = *(const v16bf*)(acf + (((c16 * 2 + 1) * 32 + lane) * 16));
    v8f accc = {0.f, 0.f, 0.f, 0.f, 0.f, 0.f, 0.f, 0.f};
    accc = __builtin_amdgcn_wmma_f32_16x16x32_bf16(false, ga0, false, bc0, (short)0, accc, false, false);
    accc = __builtin_amdgcn_wmma_f32_16x16x32_bf16(false, ga1, false, bc1, (short)0, accc, false, false);

    float addn = p2n[off + nn] + bnl[off + nn];
    float addc = p2c[off + nn] + bcl[off + nn];
    float cw2v = cw2l[off + nn];
    float sumn = 0.f;
#pragma unroll
    for (int i = 0; i < 8; ++i) {
      int row = i + 8 * ha;                               // C layout: M = i + 8*(lane>>4)
      float hn = accn[i] + p1n[row][off + nn] + addn;
      sumn += maskr[i] * silu_f(hn);
      float hc = accc[i] + p1c[row][off + nn] + addc;
      cwacc[i] += silu_f(hc) * cw2v;
    }
    sumn += __shfl_xor(sumn, 16, 32);                     // combine row halves
    if (lane < 16) pN[off + lane] = sumn;                 // hidden-domain partial
  }

  // ---- coord path: full row sums of silu(Hc)·cw2, then masked dir_vec agg ----
#pragma unroll
  for (int i = 0; i < 8; ++i) {
    float v = cwacc[i];
    v += __shfl_xor(v, 1, 32);
    v += __shfl_xor(v, 2, 32);
    v += __shfl_xor(v, 4, 32);
    v += __shfl_xor(v, 8, 32);
    cwacc[i] = v;
  }
  float xd0 = xcur[((size_t)b * NN + d) * 3 + 0];
  float xd1 = xcur[((size_t)b * NN + d) * 3 + 1];
  float xd2 = xcur[((size_t)b * NN + d) * 3 + 2];
  float t0 = 0.f, t1 = 0.f, t2 = 0.f;
#pragma unroll
  for (int i = 0; i < 8; ++i) {
    int s = s_base + i + 8 * ha;
    float mc = maskr[i] * cwacc[i];
    const float* xs = xcur + ((size_t)b * NN + s) * 3;
    t0 += mc * (xs[0] - xd0);
    t1 += mc * (xs[1] - xd1);
    t2 += mc * (xs[2] - xd2);
  }
  t0 += __shfl_xor(t0, 16, 32);
  t1 += __shfl_xor(t1, 16, 32);
  t2 += __shfl_xor(t2, 16, 32);
  if (lane == 0) {
    float* pC = partC + (((size_t)b * NT + sx) * NN + d) * 3;
    pC[0] = t0; pC[1] = t1; pC[2] = t2;
  }
}

// ----------------- deterministic reductions + layer updates -----------------
__global__ void k_reduceA(const float* __restrict__ partN, float* __restrict__ aggN) {
  int t = blockIdx.x * blockDim.x + threadIdx.x;         // B*N*256
  if (t >= BATCH * NN * HIDDEN) return;
  int hh = t & 255; int dnode = (t >> 8) % NN; int b = (t >> 8) / NN;
  float a = 0.f;
  for (int st = 0; st < NT; ++st)
    a += partN[(((size_t)b * NT + st) * NN + dnode) * HIDDEN + hh];
  aggN[((size_t)b * NN + dnode) * HIDDEN + hh] = a;
}

__global__ void k_reduceB(int l, const float* __restrict__ aggN,
                          const float* __restrict__ nw2, const float* __restrict__ nb2,
                          const float* __restrict__ cnt, float* __restrict__ hcur) {
  int t = blockIdx.x * blockDim.x + threadIdx.x;         // B*N*128
  if (t >= BATCH * NN * NODE_DIM) return;
  int o = t & 127; int dnode = (t >> 7) % NN; int b = (t >> 7) / NN;
  const float* ar = aggN + ((size_t)b * NN + dnode) * HIDDEN;
  const float* wr = nw2 + (size_t)(l * NODE_DIM + o) * HIDDEN;
  float a = cnt[b * NN + dnode] * nb2[l * NODE_DIM + o];
  for (int hh = 0; hh < HIDDEN; ++hh) a += ar[hh] * wr[hh];
  hcur[((size_t)b * NN + dnode) * NODE_DIM + o] += a;
}

__global__ void k_reduceC(const float* __restrict__ partC, float* __restrict__ xcur) {
  int t = blockIdx.x * blockDim.x + threadIdx.x;         // B*N*3
  if (t >= BATCH * NN * 3) return;
  int c = t % 3; int dnode = (t / 3) % NN; int b = t / (3 * NN);
  float a = 0.f;
  for (int st = 0; st < NT; ++st)
    a += partC[(((size_t)b * NT + st) * NN + dnode) * 3 + c];
  xcur[t] += a;
}

__global__ void k_out(const float* __restrict__ xcur, float* __restrict__ out) {
  int t = blockIdx.x * blockDim.x + threadIdx.x;
  if (t < BATCH * NN * 3) out[t] = xcur[t];
}

// ---------------------------------------------------------------------------
extern "C" void kernel_launch(void* const* d_in, const int* in_sizes, int n_in,
                              void* d_out, int out_size, void* d_ws, size_t ws_size,
                              hipStream_t stream) {
  const float* h    = (const float*)d_in[0];
  const float* iw   = (const float*)d_in[1];
  const float* ib   = (const float*)d_in[2];
  const float* ew1  = (const float*)d_in[3];
  const float* eb1  = (const float*)d_in[4];
  const float* ew2  = (const float*)d_in[5];
  const float* eb2  = (const float*)d_in[6];
  const float* nw1  = (const float*)d_in[7];
  const float* nb1  = (const float*)d_in[8];
  const float* nw2  = (const float*)d_in[9];
  const float* nb2  = (const float*)d_in[10];
  const float* cw1  = (const float*)d_in[11];
  const float* cb1  = (const float*)d_in[12];
  const float* cw2  = (const float*)d_in[13];

  float* ws = (float*)d_ws;
  float* xcur  = ws + OFF_XCUR;
  float* hcur  = ws + OFF_HCUR;
  float* Dm    = ws + OFF_D;
  float* Mm    = ws + OFF_MASK;
  float* cnt   = ws + OFF_CNT;
  float* pre1n = ws + OFF_PRE1N;
  float* pre2n = ws + OFF_PRE2N;
  float* pre1c = ws + OFF_PRE1C;
  float* pre2c = ws + OFF_PRE2C;
  float* biasn = ws + OFF_BIASN;
  float* biasc = ws + OFF_BIASC;
  float* aggN  = ws + OFF_AGGN;
  float* partN = ws + OFF_PARTN;
  float* partC = ws + OFF_PARTC;
  size_t bf_off = ((size_t)F_TOTAL * 4 + 255) & ~(size_t)255;
  unsigned short* anf = (unsigned short*)((char*)d_ws + bf_off);
  unsigned short* acf = anf + (size_t)LAYERS * 16384;

  k_prep<<<512, 256, 0, stream>>>(ew2, eb2, nw1, nb1, cw1, cb1, anf, acf, biasn, biasc);
  k_init<<<3, 128, 0, stream>>>(h, iw, ib, xcur, hcur);
  k_dist<<<288, 256, 0, stream>>>(xcur, Dm, Mm);
  k_cnt<<<3, 128, 0, stream>>>(Mm, cnt);

  for (int l = 0; l < LAYERS; ++l) {
    k_pre<<<1536, 256, 0, stream>>>(l, hcur, nw1, cw1, pre1n, pre2n, pre1c, pre2c);
    dim3 grid(NT, NT, BATCH);
    k_pair<<<grid, 512, 0, stream>>>(l, Dm, Mm, xcur, pre1n, pre2n, pre1c, pre2c,
                                     biasn, biasc, anf, acf, ew1, eb1, cw2,
                                     partN, partC);
    k_reduceA<<<384, 256, 0, stream>>>(partN, aggN);
    k_reduceB<<<192, 256, 0, stream>>>(l, aggN, nw2, nb2, cnt, hcur);
    k_reduceC<<<9, 128, 0, stream>>>(partC, xcur);
  }
  k_out<<<9, 128, 0, stream>>>(xcur, (float*)d_out);
}
